// HierarchicalStageMoE_63178968924522
// MI455X (gfx1250) — matrix operand order repaired and checked by reading, not compiled, not run on hardware
//
#include <hip/hip_runtime.h>
#include <hip/hip_bf16.h>
#include <math.h>

typedef __bf16 bf16_t;
typedef __attribute__((ext_vector_type(16))) __bf16 v16bf;
typedef __attribute__((ext_vector_type(8)))  float  v8f;

#define NTOK   8192      // B*T
#define DH     1024      // D
#define LN_EPS 1e-5f
#define LDA_ABUF 1664    // 1024 hnorm + 128 sfe + 4*128 bemb

static __device__ __forceinline__ float gelu_exact(float x) {
  return 0.5f * x * (1.0f + erff(x * 0.70710678118654752440f));
}

// ---------------- WMMA fragment loaders (ISA 7.12.2), aligned fast path only ----
// A (16x32): lanes 0-15 -> K 0-7 & 16-23, lanes 16-31 -> K 8-15 & 24-31.
// caller passes row pointer already offset by (k + hi*8).
static __device__ __forceinline__ v16bf frag_a(const bf16_t* __restrict__ p) {
  v16bf f;
  ((uint4*)&f)[0] = *(const uint4*)(p);
  ((uint4*)&f)[1] = *(const uint4*)(p + 16);
  return f;
}
// B (32x16): lanes 0-15 -> K 0-7 & 16-23, lanes 16-31 -> K 8-15 & 24-31 of column N=lane%16.
// Source is B^T [Npad][Kpad]; pointer offset by (k + hi*8).
static __device__ __forceinline__ v16bf frag_b(const bf16_t* __restrict__ p) {
  v16bf f;
  ((uint4*)&f)[0] = *(const uint4*)(p);
  ((uint4*)&f)[1] = *(const uint4*)(p + 16);
  return f;
}

// ---------------- generic bf16 WMMA GEMM, M fixed = 8192 ----------------
// block = 256 threads = 8 waves; block tile = 128(M) x 64(N); wave tile 16x64.
// A = [piece0 | piece1] along K (both stride lda, K0 % 32 == 0, K % 32 == 0).
// MODE 0: C=f32 store (+bias)           (router logits)
// MODE 1: C=bf16 store (+bias)          (stage / bundle embeddings)
// MODE 2: GELU -> bf16 (+bias)          (router hidden layers)
// MODE 3: GELU * rowscale -> bf16       (expert W1, gate weight folded in)
// MODE 4: combine: + sum_e gw*b2, write delta and hidden+alpha*delta
template <int MODE>
__global__ __launch_bounds__(256)
void wmma_gemm_kernel(const bf16_t* __restrict__ A0, const bf16_t* __restrict__ A1,
                      long lda, int K0, int K,
                      const bf16_t* __restrict__ Bt, long ldb,
                      const float* __restrict__ bias, int Nout,
                      bf16_t* __restrict__ outb, float* __restrict__ outf, long ldo,
                      const float* __restrict__ rowscale, int rs_stride, int rs_idx,
                      const float* __restrict__ gw, const float* __restrict__ b2,
                      const float* __restrict__ hidden, const float* __restrict__ alpha,
                      float* __restrict__ out2)
{
  const int lane  = threadIdx.x & 31;
  const int wid   = threadIdx.x >> 5;
  const int hi    = lane >> 4;       // lane-half
  const int lr    = lane & 15;
  const int mbase = blockIdx.y * 128 + wid * 16;
  const int nbase = blockIdx.x * 64;

  v8f acc0 = {}, acc1 = {}, acc2 = {}, acc3 = {};

  const bf16_t* arow0 = A0 + (long)(mbase + lr) * lda + hi * 8;
  const bf16_t* arow1 = A1 + (long)(mbase + lr) * lda + hi * 8;
  const bf16_t* brow0 = Bt + (long)(nbase + lr) * ldb + hi * 8;
  const bf16_t* brow1 = brow0 + 16 * ldb;
  const bf16_t* brow2 = brow0 + 32 * ldb;
  const bf16_t* brow3 = brow0 + 48 * ldb;

#define KSTEP(AROW, koff, kb)                                                          \
  {                                                                                    \
    v16bf a  = frag_a((AROW) + (koff));                                                \
    v16bf f0 = frag_b(brow0 + (kb));                                                   \
    v16bf f1 = frag_b(brow1 + (kb));                                                   \
    v16bf f2 = frag_b(brow2 + (kb));                                                   \
    v16bf f3 = frag_b(brow3 + (kb));                                                   \
    __builtin_prefetch((AROW) + (koff) + 256, 0, 1);                                   \
    acc0 = __builtin_amdgcn_wmma_f32_16x16x32_bf16(false, a, false, f0, (short)0, acc0, false, false); \
    acc1 = __builtin_amdgcn_wmma_f32_16x16x32_bf16(false, a, false, f1, (short)0, acc1, false, false); \
    acc2 = __builtin_amdgcn_wmma_f32_16x16x32_bf16(false, a, false, f2, (short)0, acc2, false, false); \
    acc3 = __builtin_amdgcn_wmma_f32_16x16x32_bf16(false, a, false, f3, (short)0, acc3, false, false); \
  }

#pragma unroll 2
  for (int k = 0; k < K0; k += 32) KSTEP(arow0, k, k)
#pragma unroll 2
  for (int k = K0; k < K; k += 32) KSTEP(arow1, k - K0, k)
#undef KSTEP

  // C/D layout: col = lane%16, row = r + 8*(lane/16)
  v8f accs[4] = {acc0, acc1, acc2, acc3};
#pragma unroll
  for (int t = 0; t < 4; ++t) {
    const int n = nbase + 16 * t + lr;
    if (n >= Nout) continue;
    const float bv = bias ? bias[n] : 0.0f;
#pragma unroll
    for (int r = 0; r < 8; ++r) {
      const int m = mbase + hi * 8 + r;
      float v = accs[t][r] + bv;
      if (MODE == 2 || MODE == 3) v = gelu_exact(v);
      if (MODE == 3) v *= rowscale[(long)m * rs_stride + rs_idx];
      if (MODE == 0) {
        outf[(long)m * ldo + n] = v;
      } else if (MODE == 1 || MODE == 2 || MODE == 3) {
        outb[(long)m * ldo + n] = (bf16_t)v;
      } else {  // MODE 4: final combine
        float bm = 0.0f;
#pragma unroll
        for (int e = 0; e < 16; ++e) bm += gw[(long)m * 16 + e] * b2[e * DH + n];
        v += bm;
        outf[(long)m * DH + n] = v;                                        // stage_delta
        out2[(long)m * DH + n] = hidden[(long)m * DH + n] + alpha[0] * v;  // next_hidden
      }
    }
  }
}

// ---------------- LayerNorm -> bf16 into Abuf cols [0,1024) ----------------
__global__ __launch_bounds__(256)
void ln_cast_kernel(const float* __restrict__ hidden, const float* __restrict__ g,
                    const float* __restrict__ b, bf16_t* __restrict__ Abuf)
{
  const int t = blockIdx.x;
  const float* h = hidden + (long)t * DH;
  float v[4]; float s = 0.f, s2 = 0.f;
#pragma unroll
  for (int i = 0; i < 4; ++i) {
    v[i] = h[threadIdx.x + 256 * i]; s += v[i]; s2 += v[i] * v[i];
  }
  __shared__ float rs[256], rq[256];
  rs[threadIdx.x] = s; rq[threadIdx.x] = s2;
  __syncthreads();
  for (int off = 128; off > 0; off >>= 1) {
    if (threadIdx.x < off) {
      rs[threadIdx.x] += rs[threadIdx.x + off];
      rq[threadIdx.x] += rq[threadIdx.x + off];
    }
    __syncthreads();
  }
  const float mu   = rs[0] * (1.0f / DH);
  const float var  = rq[0] * (1.0f / DH) - mu * mu;
  const float rstd = rsqrtf(var + LN_EPS);
  bf16_t* row = Abuf + (long)t * LDA_ABUF;
#pragma unroll
  for (int i = 0; i < 4; ++i) {
    const int c = threadIdx.x + 256 * i;
    row[c] = (bf16_t)((v[i] - mu) * rstd * g[c] + b[c]);
  }
}

// ------- weight cast+transpose: f32 [K][Nsrc] -> bf16 [Npad][Kpad], zero padded -------
__global__ void cast_transpose_kernel(const float* __restrict__ src, bf16_t* __restrict__ dst,
                                      int K, int Kpad, int Nsrc, int Npad,
                                      long sbatch, long dbatch)
{
  src += (long)blockIdx.z * sbatch;
  dst += (long)blockIdx.z * dbatch;
  __shared__ float tile[32][33];
  const int k0 = blockIdx.y * 32, n0 = blockIdx.x * 32;
  for (int i = threadIdx.y; i < 32; i += 8) {
    const int k = k0 + i, n = n0 + threadIdx.x;
    tile[i][threadIdx.x] = (k < K && n < Nsrc) ? src[(long)k * Nsrc + n] : 0.0f;
  }
  __syncthreads();
  for (int i = threadIdx.y; i < 32; i += 8) {
    const int n = n0 + i, k = k0 + threadIdx.x;
    if (n < Npad && k < Kpad) dst[(long)n * Kpad + k] = (bf16_t)tile[threadIdx.x][i];
  }
}

__global__ void cast_bf16_kernel(const float* __restrict__ src, bf16_t* __restrict__ dst, long n) {
  const long i = (long)blockIdx.x * 256 + threadIdx.x;
  if (i < n) dst[i] = (bf16_t)src[i];
}

// bundle features packed+padded: featp[t][g][j] (j<32), j>=16 -> 0
__global__ void bemb_pack_kernel(const float* __restrict__ feat, bf16_t* __restrict__ featp) {
  const long i = (long)blockIdx.x * 256 + threadIdx.x;
  if (i >= (long)NTOK * 128) return;
  const int  c = (int)(i & 127);
  const long t = i >> 7;
  const int  g = c >> 5, j = c & 31;
  const float v = (j < 16) ? feat[t * 64 + g * 16 + j] : 0.0f;
  featp[i] = (bf16_t)v;
}

// ---------------- gating: top-2 softmax over 4 (bundles) and per-bundle 4 ----------------
static __device__ __forceinline__ void top2_softmax4(const float* s, float* w) {
  float m1 = -3.4e38f, m2 = -3.4e38f;
#pragma unroll
  for (int i = 0; i < 4; ++i) {
    const float v = s[i];
    if (v > m1) { m2 = m1; m1 = v; } else if (v > m2) m2 = v;
  }
  float e[4]; float sum = 0.f;
#pragma unroll
  for (int i = 0; i < 4; ++i) { e[i] = (s[i] >= m2) ? __expf(s[i] - m1) : 0.0f; sum += e[i]; }
  const float inv = 1.0f / sum;
#pragma unroll
  for (int i = 0; i < 4; ++i) w[i] = e[i] * inv;
}

__global__ void gate_kernel(const float* __restrict__ bl, const float* __restrict__ il,
                            float* __restrict__ o_bw, float* __restrict__ o_gw,
                            float* __restrict__ o_gl)
{
  const int t = blockIdx.x * 256 + threadIdx.x;
  if (t >= NTOK) return;
  float b[4];
#pragma unroll
  for (int g = 0; g < 4; ++g) b[g] = bl[t * 4 + g];
  float bw[4]; top2_softmax4(b, bw);
#pragma unroll
  for (int g = 0; g < 4; ++g) o_bw[t * 4 + g] = bw[g];
  float l[16];
#pragma unroll
  for (int i = 0; i < 16; ++i) l[i] = il[t * 16 + i];
#pragma unroll
  for (int g = 0; g < 4; ++g) {
    float iw[4]; top2_softmax4(l + 4 * g, iw);
#pragma unroll
    for (int j = 0; j < 4; ++j) {
      o_gw[t * 16 + 4 * g + j] = bw[g] * iw[j];
      o_gl[t * 16 + 4 * g + j] = b[g] + l[4 * g + j];
    }
  }
}

// ---------------- host launch ----------------
#define GEMM(MODE, A0_, A1_, lda_, K0_, K_, Bt_, ldb_, bias_, Nout_, outb_, outf_, ldo_, \
             rs_, rss_, rsi_, gw_, b2_, hid_, al_, out2_)                                \
  wmma_gemm_kernel<MODE><<<dim3(((Nout_) + 63) / 64, NTOK / 128, 1), 256, 0, stream>>>(  \
      A0_, A1_, (long)(lda_), K0_, K_, Bt_, (long)(ldb_), bias_, Nout_, outb_, outf_,    \
      (long)(ldo_), rs_, rss_, rsi_, gw_, b2_, hid_, al_, out2_)

extern "C" void kernel_launch(void* const* d_in, const int* in_sizes, int n_in,
                              void* d_out, int out_size, void* d_ws, size_t ws_size,
                              hipStream_t stream) {
  (void)in_sizes; (void)n_in; (void)out_size; (void)ws_size;
  const float* hidden  = (const float*)d_in[0];
  const float* feat    = (const float*)d_in[1];
  const float* ln_g    = (const float*)d_in[2];
  const float* ln_b    = (const float*)d_in[3];
  const float* stage_W = (const float*)d_in[4];
  const float* stage_b = (const float*)d_in[5];
  const float* bproj_W = (const float*)d_in[6];
  const float* bproj_b = (const float*)d_in[7];
  const float* br_W1   = (const float*)d_in[8];
  const float* br_b1   = (const float*)d_in[9];
  const float* br_W2   = (const float*)d_in[10];
  const float* br_b2   = (const float*)d_in[11];
  const float* ir_W1   = (const float*)d_in[12];
  const float* ir_b1   = (const float*)d_in[13];
  const float* ir_W2   = (const float*)d_in[14];
  const float* ir_b2   = (const float*)d_in[15];
  const float* ex_W1   = (const float*)d_in[16];
  const float* ex_b1   = (const float*)d_in[17];
  const float* ex_W2   = (const float*)d_in[18];
  const float* ex_b2   = (const float*)d_in[19];
  const float* alpha   = (const float*)d_in[20];

  float* out      = (float*)d_out;
  float* o_next   = out;                 // [8192,1024]
  float* o_gw     = out + 8388608;       // [8192,16]
  float* o_gl     = o_gw + 131072;       // [8192,16]
  float* o_bw     = o_gl + 131072;       // [8192,4]
  float* o_bl     = o_bw + 32768;        // [8192,4]
  float* o_delta  = o_bl + 32768;        // [8192,1024]

  // workspace carve
  uintptr_t base = (uintptr_t)d_ws;
  size_t off = 0;
  auto carve = [&](size_t bytes) -> void* {
    off = (off + 255) & ~(size_t)255;
    void* p = (void*)(base + off);
    off += bytes;
    return p;
  };
  bf16_t* Abuf   = (bf16_t*)carve((size_t)NTOK * LDA_ABUF * 2);   // [hnorm|sfe|bemb*4]
  bf16_t* featb  = (bf16_t*)carve((size_t)NTOK * 64 * 2);         // features, K=64
  bf16_t* featp  = (bf16_t*)carve((size_t)NTOK * 128 * 2);        // per-bundle, K padded 16->32
  bf16_t* rhb    = (bf16_t*)carve((size_t)NTOK * 256 * 2);        // bundle router hidden
  bf16_t* rhi    = (bf16_t*)carve((size_t)NTOK * 256 * 2);        // intra router hidden (reused)
  float*  intr   = (float*)carve((size_t)NTOK * 16 * 4);          // intra logits [N,16]
  bf16_t* ehs    = (bf16_t*)carve((size_t)NTOK * 8192 * 2);       // gate*GELU(eh), [N,16*512]
  bf16_t* stageWt = (bf16_t*)carve((size_t)128 * 64 * 2);
  bf16_t* bprojt  = (bf16_t*)carve((size_t)4 * 128 * 32 * 2);     // K padded to 32
  bf16_t* brW1t   = (bf16_t*)carve((size_t)256 * 1152 * 2);
  bf16_t* brW2t   = (bf16_t*)carve((size_t)64 * 256 * 2);
  bf16_t* irW1t   = (bf16_t*)carve((size_t)4 * 256 * 1152 * 2);
  bf16_t* irW2t   = (bf16_t*)carve((size_t)4 * 64 * 256 * 2);
  bf16_t* exW1t   = (bf16_t*)carve((size_t)16 * 512 * 1024 * 2);
  bf16_t* exW2t   = (bf16_t*)carve((size_t)1024 * 8192 * 2);

  // ---- weight prep: cast + transpose (N pad -> mult of 64, K pad -> mult of 32) ----
  dim3 tb(32, 8, 1);
  auto tl = [&](const float* s, bf16_t* d, int K, int Kpad, int Nsrc, int Npad, int batch) {
    dim3 grid((Npad + 31) / 32, (Kpad + 31) / 32, batch);
    cast_transpose_kernel<<<grid, tb, 0, stream>>>(s, d, K, Kpad, Nsrc, Npad,
                                                   (long)K * Nsrc, (long)Npad * Kpad);
  };
  tl(stage_W, stageWt, 64, 64, 128, 128, 1);
  tl(bproj_W, bprojt, 16, 32, 128, 128, 4);
  tl(br_W1, brW1t, 1152, 1152, 256, 256, 1);
  tl(br_W2, brW2t, 256, 256, 4, 64, 1);
  tl(ir_W1, irW1t, 1152, 1152, 256, 256, 4);
  tl(ir_W2, irW2t, 256, 256, 4, 64, 4);
  tl(ex_W1, exW1t, 1024, 1024, 512, 512, 16);
  tl(ex_W2, exW2t, 8192, 8192, 1024, 1024, 1);   // stacked [16*512, 1024]

  // ---- activations ----
  cast_bf16_kernel<<<(NTOK * 64 + 255) / 256, 256, 0, stream>>>(feat, featb, (long)NTOK * 64);
  bemb_pack_kernel<<<(NTOK * 128 + 255) / 256, 256, 0, stream>>>(feat, featp);
  ln_cast_kernel<<<NTOK, 256, 0, stream>>>(hidden, ln_g, ln_b, Abuf);

  const float* FN = nullptr; bf16_t* BN = nullptr; float* PN = nullptr;

  // stage feature embedding -> Abuf cols [1024,1152)
  GEMM(1, featb, featb, 64, 64, 64, stageWt, 64, stage_b, 128,
       Abuf + 1024, PN, LDA_ABUF, FN, 0, 0, FN, FN, FN, FN, PN);
  // bundle embeddings -> Abuf cols [1152+128g, ...)
  for (int g = 0; g < 4; ++g)
    GEMM(1, featp + 32 * g, featp + 32 * g, 128, 32, 32, bprojt + (size_t)g * 128 * 32, 32,
         bproj_b + 128 * g, 128, Abuf + 1152 + 128 * g, PN, LDA_ABUF,
         FN, 0, 0, FN, FN, FN, FN, PN);
  // bundle router: hidden (GELU) then logits
  GEMM(2, Abuf, Abuf, LDA_ABUF, 1152, 1152, brW1t, 1152, br_b1, 256,
       rhb, PN, 256, FN, 0, 0, FN, FN, FN, FN, PN);
  GEMM(0, rhb, rhb, 256, 256, 256, brW2t, 256, br_b2, 4,
       BN, o_bl, 4, FN, 0, 0, FN, FN, FN, FN, PN);
  // intra routers (two-piece K: hnorm cols + this bundle's embedding cols)
  for (int g = 0; g < 4; ++g) {
    GEMM(2, Abuf, Abuf + 1152 + 128 * g, LDA_ABUF, 1024, 1152,
         irW1t + (size_t)g * 256 * 1152, 1152, ir_b1 + 256 * g, 256,
         rhi, PN, 256, FN, 0, 0, FN, FN, FN, FN, PN);
    GEMM(0, rhi, rhi, 256, 256, 256, irW2t + (size_t)g * 64 * 256, 256,
         ir_b2 + 4 * g, 4, BN, intr + 4 * g, 16, FN, 0, 0, FN, FN, FN, FN, PN);
  }
  // gating (top-2 softmax x2) -> bundle_weights, gate_weights, gate_logits
  gate_kernel<<<NTOK / 256, 256, 0, stream>>>(o_bl, intr, o_bw, o_gw, o_gl);

  // experts W1: eh = gate * GELU(hnorm @ W1 + b1), bf16 into stacked K buffer
  for (int e = 0; e < 16; ++e)
    GEMM(3, Abuf, Abuf, LDA_ABUF, 1024, 1024, exW1t + (size_t)e * 512 * 1024, 1024,
         ex_b1 + 512 * e, 512, ehs + 512 * e, PN, 8192,
         o_gw, 16, e, FN, FN, FN, FN, PN);

  // combine: delta = ehs @ stacked(W2) + sum_e g_e*b2_e ; next = hidden + alpha*delta
  GEMM(4, ehs, ehs, 8192, 8192, 8192, exW2t, 8192, FN, 1024,
       BN, o_delta, 1024, FN, 0, 0, o_gw, ex_b2, hidden, alpha, o_next);
}
